// PosetBlock_86921548136533
// MI455X (gfx1250) — compile-verified, matrix operand order checked by reading, not run on hardware
//
#include <hip/hip_runtime.h>
#include <hip/hip_bf16.h>
#include <math.h>

// ---------------- problem constants ----------------
#define BB        4
#define TT        2048
#define DM        1024
#define NH        16
#define DH        64
#define BZ        (BB * NH)          // 64
#define KP        16                 // parents per token
#define INV_TAU   (1.0f / 0.07f)
#define M_ROWS    (BB * TT)          // 8192

// ---------------- workspace layout (bytes) ----------------
static const size_t OFF_XLN = 0;
static const size_t OFF_W   = OFF_XLN + (size_t)M_ROWS * DM * 2;          //  16777216
static const size_t OFF_QK  = OFF_W   + (size_t)4 * DM * DM * 2;          //  25165824
static const size_t OFF_K   = OFF_QK  + (size_t)BZ * TT * DH * 2;         //  41943040
static const size_t OFF_V   = OFF_K   + (size_t)BZ * TT * DH * 2;         //  58720256
static const size_t OFF_A   = OFF_V   + (size_t)BZ * TT * DH * 4;         //  92274688
static const size_t OFF_BS1 = OFF_A   + (size_t)BZ * TT * KP * 4;         // 100663296
static const size_t OFF_Z0  = OFF_BS1 + (size_t)BZ * TT * DH * 4;         // 134217728
static const size_t OFF_Z1  = OFF_Z0  + (size_t)BZ * TT * 4;              // 134742016

// ---------------- types ----------------
typedef __attribute__((ext_vector_type(16))) __bf16 v16bf;
typedef __attribute__((ext_vector_type(8)))  __bf16 v8bf;
typedef __attribute__((ext_vector_type(8)))  float  v8f;
typedef __attribute__((ext_vector_type(4)))  float  v4f;

__device__ __forceinline__ float bf2f(unsigned short u) {
    unsigned int x = ((unsigned int)u) << 16;
    float f; __builtin_memcpy(&f, &x, 4); return f;
}
__device__ __forceinline__ unsigned short f2bf(float f) {
    unsigned int x; __builtin_memcpy(&x, &f, 4);
    unsigned int r = x + 0x7FFFu + ((x >> 16) & 1u);   // round-to-nearest-even
    return (unsigned short)(r >> 16);
}

// Load one 16x32 bf16 WMMA operand fragment (ISA layout: lane m holds
// K=[koff..koff+7] and K=[koff+16..koff+23], koff = (lane>>4)*8).
__device__ __forceinline__ v16bf load_frag(const unsigned short* rowp, int koff) {
    union { v16bf v; v8bf h[2]; } u;
    u.h[0] = *(const v8bf*)(rowp + koff);
    u.h[1] = *(const v8bf*)(rowp + koff + 16);
    return u.v;
}

#define WMMA_BF16(A_, B_, C_) \
    __builtin_amdgcn_wmma_f32_16x16x32_bf16(false, (A_), false, (B_), (short)0, (C_), false, false)

// Shared MMA core: wave computes 32x64 C sub-tile (2x4 WMMA tiles), K = DM.
// Scalar pointers (no arrays!) advanced per K-chunk -> no scratch spills.
__device__ __forceinline__ void mma_core(const unsigned short* __restrict__ A,
                                         const unsigned short* __restrict__ W,
                                         int m0, int n0, int lane,
                                         v8f acc[2][4]) {
    const int koff = (lane >> 4) << 3;
    const int rsel = lane & 15;
    const unsigned short* pA0 = A + (size_t)(m0 +  0 + rsel) * DM;
    const unsigned short* pA1 = A + (size_t)(m0 + 16 + rsel) * DM;
    const unsigned short* pB0 = W + (size_t)(n0 +  0 + rsel) * DM;
    const unsigned short* pB1 = W + (size_t)(n0 + 16 + rsel) * DM;
    const unsigned short* pB2 = W + (size_t)(n0 + 32 + rsel) * DM;
    const unsigned short* pB3 = W + (size_t)(n0 + 48 + rsel) * DM;

    #pragma unroll
    for (int i = 0; i < 2; ++i)
        #pragma unroll
        for (int j = 0; j < 4; ++j) acc[i][j] = (v8f)(0.0f);

    #pragma unroll 4
    for (int kk = 0; kk < DM; kk += 32) {
        // speculative prefetch two K-chunks ahead (silently dropped past end)
        __builtin_prefetch(pA0 + 64, 0, 1);
        __builtin_prefetch(pB0 + 64, 0, 1);
        v16bf a0 = load_frag(pA0, koff);
        v16bf a1 = load_frag(pA1, koff);
        v16bf b0 = load_frag(pB0, koff);
        v16bf b1 = load_frag(pB1, koff);
        v16bf b2 = load_frag(pB2, koff);
        v16bf b3 = load_frag(pB3, koff);
        acc[0][0] = WMMA_BF16(a0, b0, acc[0][0]);
        acc[0][1] = WMMA_BF16(a0, b1, acc[0][1]);
        acc[0][2] = WMMA_BF16(a0, b2, acc[0][2]);
        acc[0][3] = WMMA_BF16(a0, b3, acc[0][3]);
        acc[1][0] = WMMA_BF16(a1, b0, acc[1][0]);
        acc[1][1] = WMMA_BF16(a1, b1, acc[1][1]);
        acc[1][2] = WMMA_BF16(a1, b2, acc[1][2]);
        acc[1][3] = WMMA_BF16(a1, b3, acc[1][3]);
        pA0 += 32; pA1 += 32;
        pB0 += 32; pB1 += 32; pB2 += 32; pB3 += 32;
    }
}

// ---------------- 1) LayerNorm -> bf16 ----------------
__global__ void ln_kernel(const float* __restrict__ x,
                          const float* __restrict__ g,
                          const float* __restrict__ b,
                          unsigned short* __restrict__ xln) {
    __shared__ float ws1[8], ws2[8], sMu, sRs;
    const int row  = blockIdx.x;
    const int tid  = threadIdx.x;
    const int lane = tid & 31, wid = tid >> 5;
    const size_t base = (size_t)row * DM + tid * 4;
    v4f xv = *(const v4f*)(x + base);
    float s = xv[0] + xv[1] + xv[2] + xv[3];
    float ss = xv[0]*xv[0] + xv[1]*xv[1] + xv[2]*xv[2] + xv[3]*xv[3];
    #pragma unroll
    for (int off = 16; off; off >>= 1) {
        s  += __shfl_xor(s,  off, 32);
        ss += __shfl_xor(ss, off, 32);
    }
    if (lane == 0) { ws1[wid] = s; ws2[wid] = ss; }
    __syncthreads();
    if (tid == 0) {
        float ts = 0.f, tss = 0.f;
        #pragma unroll
        for (int i = 0; i < 8; ++i) { ts += ws1[i]; tss += ws2[i]; }
        float mu  = ts * (1.0f / DM);
        float var = tss * (1.0f / DM) - mu * mu;   // biased variance
        sMu = mu; sRs = rsqrtf(var + 1e-5f);
    }
    __syncthreads();
    const float mu = sMu, rs = sRs;
    #pragma unroll
    for (int i = 0; i < 4; ++i) {
        int col = tid * 4 + i;
        float y = (xv[i] - mu) * rs * g[col] + b[col];
        xln[base + i] = f2bf(y);
    }
}

// ---------------- 2) f32 -> bf16 convert ----------------
__global__ void conv_kernel(const float* __restrict__ w,
                            unsigned short* __restrict__ o, int n) {
    int i = (blockIdx.x * blockDim.x + threadIdx.x) * 4;
    if (i + 3 < n) {
        v4f wv = *(const v4f*)(w + i);
        o[i + 0] = f2bf(wv[0]); o[i + 1] = f2bf(wv[1]);
        o[i + 2] = f2bf(wv[2]); o[i + 3] = f2bf(wv[3]);
    }
}

// ---------------- 3) QKV GEMM: C = xln * W^T, scattered to [BZ,T,DH] ------
// block = 256 threads (8 waves); block tile 128x128; wave tile 32x64.
__global__ void gemm_qkv_kernel(const unsigned short* __restrict__ A,
                                const unsigned short* __restrict__ Wqkv,
                                unsigned short* __restrict__ qb,
                                unsigned short* __restrict__ kb,
                                float* __restrict__ vf) {
    const int which = blockIdx.z;                       // 0=Q 1=K 2=V
    const unsigned short* W = Wqkv + (size_t)which * DM * DM;
    const int lane = threadIdx.x & 31, wid = threadIdx.x >> 5;
    const int waveM = wid >> 1, waveN = wid & 1;
    const int m0 = blockIdx.y * 128 + waveM * 32;
    const int n0 = blockIdx.x * 128 + waveN * 64;

    v8f acc[2][4];
    mma_core(A, W, m0, n0, lane, acc);

    // epilogue: C layout VGPR r -> M = r + (lane>>4)*8, N = lane&15
    const int mloc = (lane >> 4) << 3;
    const int rsel = lane & 15;
    if (which == 0) {
        #pragma unroll
        for (int i = 0; i < 2; ++i)
            #pragma unroll
            for (int j = 0; j < 4; ++j)
                #pragma unroll
                for (int r = 0; r < 8; ++r) {
                    int m = m0 + i * 16 + mloc + r, n = n0 + j * 16 + rsel;
                    int bbat = m >> 11, t = m & (TT - 1), hh = n >> 6, c = n & (DH - 1);
                    qb[(((size_t)(bbat * NH + hh)) * TT + t) * DH + c] = f2bf(acc[i][j][r]);
                }
    } else if (which == 1) {
        #pragma unroll
        for (int i = 0; i < 2; ++i)
            #pragma unroll
            for (int j = 0; j < 4; ++j)
                #pragma unroll
                for (int r = 0; r < 8; ++r) {
                    int m = m0 + i * 16 + mloc + r, n = n0 + j * 16 + rsel;
                    int bbat = m >> 11, t = m & (TT - 1), hh = n >> 6, c = n & (DH - 1);
                    kb[(((size_t)(bbat * NH + hh)) * TT + t) * DH + c] = f2bf(acc[i][j][r]);
                }
    } else {
        #pragma unroll
        for (int i = 0; i < 2; ++i)
            #pragma unroll
            for (int j = 0; j < 4; ++j)
                #pragma unroll
                for (int r = 0; r < 8; ++r) {
                    int m = m0 + i * 16 + mloc + r, n = n0 + j * 16 + rsel;
                    int bbat = m >> 11, t = m & (TT - 1), hh = n >> 6, c = n & (DH - 1);
                    vf[(((size_t)(bbat * NH + hh)) * TT + t) * DH + c] = acc[i][j][r];
                }
    }
}

// ---------------- 4) edge gates: a[z,t,j] -------------------------------
// one wave per (z,t); lane l covers dh dims {l, l+32}; 16 parents.
__global__ void gates_kernel(const unsigned short* __restrict__ qb,
                             const unsigned short* __restrict__ kb,
                             const float* __restrict__ relb,
                             float* __restrict__ a) {
    const int lane = threadIdx.x & 31, wid = threadIdx.x >> 5;
    const int gid = blockIdx.x * 8 + wid;
    const int z = gid >> 11, t = gid & (TT - 1);
    const unsigned short* qp = qb + ((size_t)z * TT + t) * DH;
    const float q0 = bf2f(qp[lane]), q1 = bf2f(qp[lane + 32]);
    float myA = 0.0f;
    #pragma unroll
    for (int j = 0; j < KP; ++j) {
        const int s = t - KP + j;
        float aval = 0.0f;
        if (s >= 0) {
            const unsigned short* kp = kb + ((size_t)z * TT + s) * DH;
            float p = q0 * bf2f(kp[lane]) + q1 * bf2f(kp[lane + 32]);
            #pragma unroll
            for (int off = 16; off; off >>= 1) p += __shfl_xor(p, off, 32);
            float logit = p * 0.125f + relb[t - s];     // 1/sqrt(64); delta<=16<WINDOW
            float I = 1.0f / (1.0f + __expf(-logit));
            I = fminf(fmaxf(I, 1e-6f), 1.0f - 1e-6f);
            aval = __expf(__logf(I) * INV_TAU);
        }
        if (lane == j) myA = aval;
    }
    if (lane < KP) a[((size_t)z * TT + t) * KP + lane] = myA;
}

// ---------------- 5) poset aggregation round ----------------------------
__global__ void agg_kernel(const float* __restrict__ a,
                           const float* __restrict__ v,
                           const float* __restrict__ Bsin,
                           const float* __restrict__ Zin,
                           float* __restrict__ Bsout,
                           float* __restrict__ Zout,
                           unsigned short* __restrict__ hout,
                           int firstRound, int finalRound) {
    const int lane = threadIdx.x & 31, wid = threadIdx.x >> 5;
    const int gid = blockIdx.x * 8 + wid;
    const int z = gid >> 11, t = gid & (TT - 1);
    float sb0 = 0.f, sb1 = 0.f, sz = 0.f;
    for (int j = 0; j < KP; ++j) {
        const int s = t - KP + j;
        if (s < 0) continue;
        const float aj = a[((size_t)z * TT + t) * KP + j];
        const size_t off = ((size_t)z * TT + s) * DH;
        sb0 += aj * Bsin[off + lane];
        sb1 += aj * Bsin[off + lane + 32];
        sz  += aj * (firstRound ? 1.0f : Zin[(size_t)z * TT + s]);
    }
    const size_t oo = ((size_t)z * TT + t) * DH;
    if (!finalRound) {
        Bsout[oo + lane]      = v[oo + lane]      + sb0;
        Bsout[oo + lane + 32] = v[oo + lane + 32] + sb1;
        if (lane == 0) Zout[(size_t)z * TT + t] = 1.0f + sz;
    } else {
        const float inv = 1.0f / fmaxf(sz, 1e-9f);
        const int bbat = z >> 4, hh = z & 15;
        const size_t o = ((size_t)(bbat * TT + t)) * DM + hh * DH;
        hout[o + lane]      = f2bf(sb0 * inv);
        hout[o + lane + 32] = f2bf(sb1 * inv);
    }
}

// ---------------- 6) output GEMM: out = x + h * Wo^T --------------------
__global__ void gemm_out_kernel(const unsigned short* __restrict__ A,   // h bf16
                                const unsigned short* __restrict__ W,   // Wo bf16
                                const float* __restrict__ x,
                                float* __restrict__ out) {
    const int lane = threadIdx.x & 31, wid = threadIdx.x >> 5;
    const int waveM = wid >> 1, waveN = wid & 1;
    const int m0 = blockIdx.y * 128 + waveM * 32;
    const int n0 = blockIdx.x * 128 + waveN * 64;

    v8f acc[2][4];
    mma_core(A, W, m0, n0, lane, acc);

    const int mloc = (lane >> 4) << 3;
    const int rsel = lane & 15;
    #pragma unroll
    for (int i = 0; i < 2; ++i)
        #pragma unroll
        for (int j = 0; j < 4; ++j)
            #pragma unroll
            for (int r = 0; r < 8; ++r) {
                int m = m0 + i * 16 + mloc + r;
                int n = n0 + j * 16 + rsel;
                size_t idx = (size_t)m * DM + n;
                out[idx] = x[idx] + acc[i][j][r];
            }
}

// ---------------- host-side launch ----------------
extern "C" void kernel_launch(void* const* d_in, const int* in_sizes, int n_in,
                              void* d_out, int out_size, void* d_ws, size_t ws_size,
                              hipStream_t stream) {
    const float* x    = (const float*)d_in[0];
    const float* ln_g = (const float*)d_in[1];
    const float* ln_b = (const float*)d_in[2];
    const float* Wq   = (const float*)d_in[3];
    const float* Wk   = (const float*)d_in[4];
    const float* Wv   = (const float*)d_in[5];
    const float* Wo   = (const float*)d_in[6];
    const float* relb = (const float*)d_in[7];
    float* out = (float*)d_out;

    char* ws = (char*)d_ws;
    unsigned short* xln = (unsigned short*)(ws + OFF_XLN);   // also h bf16 later
    unsigned short* wbf = (unsigned short*)(ws + OFF_W);     // Wq,Wk,Wv,Wo bf16
    unsigned short* qb  = (unsigned short*)(ws + OFF_QK);
    unsigned short* kb  = (unsigned short*)(ws + OFF_K);
    float* vf  = (float*)(ws + OFF_V);
    float* af  = (float*)(ws + OFF_A);
    float* Bs0 = (float*)(ws + OFF_QK);                      // aliases q|k (dead then)
    float* Bs1 = (float*)(ws + OFF_BS1);
    float* Z0  = (float*)(ws + OFF_Z0);
    float* Z1  = (float*)(ws + OFF_Z1);
    unsigned short* hbf = xln;                               // aliases xln (dead then)

    // 1) LayerNorm
    ln_kernel<<<M_ROWS, 256, 0, stream>>>(x, ln_g, ln_b, xln);

    // 2) weight conversions (f32 -> bf16)
    const int wn = DM * DM;                                  // 1048576
    const int cb = (wn / 4 + 255) / 256;                     // 1024 blocks
    conv_kernel<<<cb, 256, 0, stream>>>(Wq, wbf + 0 * (size_t)wn, wn);
    conv_kernel<<<cb, 256, 0, stream>>>(Wk, wbf + 1 * (size_t)wn, wn);
    conv_kernel<<<cb, 256, 0, stream>>>(Wv, wbf + 2 * (size_t)wn, wn);
    conv_kernel<<<cb, 256, 0, stream>>>(Wo, wbf + 3 * (size_t)wn, wn);

    // 3) QKV projections (WMMA bf16)
    gemm_qkv_kernel<<<dim3(DM / 128, M_ROWS / 128, 3), 256, 0, stream>>>(
        xln, wbf, qb, kb, vf);

    // 4) edge gates
    gates_kernel<<<(BZ * TT) / 8, 256, 0, stream>>>(qb, kb, relb, af);

    // 5) four aggregation rounds (Bs: v -> Bs0 -> Bs1 -> Bs0 -> h)
    const int ablocks = (BZ * TT) / 8;
    agg_kernel<<<ablocks, 256, 0, stream>>>(af, vf, vf,  Z0, Bs0, Z0, hbf, 1, 0);
    agg_kernel<<<ablocks, 256, 0, stream>>>(af, vf, Bs0, Z0, Bs1, Z1, hbf, 0, 0);
    agg_kernel<<<ablocks, 256, 0, stream>>>(af, vf, Bs1, Z1, Bs0, Z0, hbf, 0, 0);
    agg_kernel<<<ablocks, 256, 0, stream>>>(af, vf, Bs0, Z0, Bs1, Z1, hbf, 0, 1);

    // 6) output GEMM + residual (WMMA bf16)
    gemm_out_kernel<<<dim3(DM / 128, M_ROWS / 128, 1), 256, 0, stream>>>(
        hbf, wbf + 3 * (size_t)wn, x, out);
}
